// GraphSinkhornPooling_36266703847507
// MI455X (gfx1250) — compile-verified, matrix operand order checked by reading, not run on hardware
//
#include <hip/hip_runtime.h>
#include <math.h>

// ---------------------------------------------------------------------------
// Graph Sinkhorn pooling for MI455X (gfx1250, wave32).
//   - K_log (32MB) is L2-resident (192MB L2): the 200 Sinkhorn sweeps are
//     L2-bandwidth + v_exp_f32 bound, not HBM bound.
//   - The one real GEMM (H[65536,128] @ d2[128,128]) uses
//     v_wmma_f32_16x16x32_f16 with pre-swizzled A/B tiles in the ISA-
//     documented 16-bit A-matrix VGPR layout (32B contiguous per lane).
// ---------------------------------------------------------------------------

#define NN      65536      // nodes
#define KK      128        // atoms
#define DD      256        // embedding dim
#define BB      128        // graphs
#define EPS_INV 10.0f      // 1/eps_sink
#define MAXIT   100
#define TINYF   1e-12f
// log(1/128 + 1e-12)
#define B_LOG   (-4.8520303f)

typedef _Float16 half16v __attribute__((ext_vector_type(16)));
typedef _Float16 half2v  __attribute__((ext_vector_type(2)));
typedef float    f8v     __attribute__((ext_vector_type(8)));

// ---------------- setup kernels ----------------

__global__ void k_init(float* v, int* cnt) {
    int idx = blockIdx.x * 256 + threadIdx.x;
    if (idx < BB * KK) v[idx] = 0.0f;
    if (idx < BB) cnt[idx] = 0;
}

__global__ void k_count(const int* __restrict__ bidx, int* cnt) {
    int i = blockIdx.x * 256 + threadIdx.x;
    if (i < NN) atomicAdd(&cnt[bidx[i]], 1);
}

__global__ void k_scan(const int* __restrict__ cnt, int* starts, float* a_log_g) {
    int t = threadIdx.x;
    if (t < BB) {
        int c = cnt[t];
        a_log_g[t] = (c > 0) ? logf(1.0f / (float)c + TINYF) : 0.0f;
    }
    if (t == 0) {
        int acc = 0;
        for (int g = 0; g < BB; ++g) { starts[g] = acc; acc += cnt[g]; }
        starts[BB] = acc;
    }
}

// d2[k][n] = |c_k - c_n|^2, stored directly in the WMMA 16-bit B layout:
// Bsw[((chunk*8 + colTile)*32 + lane)*16 + v*2 + h]
__global__ void k_d2sw(const float* __restrict__ codebook, _Float16* Bsw) {
    __shared__ float ck[DD];
    int k = blockIdx.x;            // 128 blocks
    int n = threadIdx.x;           // 128 threads
    ck[n]        = codebook[k * DD + n];
    ck[n + 128]  = codebook[k * DD + 128 + n];
    __syncthreads();
    float acc = 0.0f;
    for (int d = 0; d < DD; ++d) {
        float diff = ck[d] - codebook[n * DD + d];
        acc += diff * diff;
    }
    // swizzle (k,n) -> (chunk, colTile, lane, v, h)
    int chunk = k >> 5;
    int kin   = k & 31;
    int g     = kin >> 3;            // 0..3
    int laneHi = g & 1;
    int vbase  = (g >> 1) * 4;
    int within = kin & 7;
    int v      = vbase + (within >> 1);
    int h      = within & 1;
    int lane   = (n & 15) | (laneHi << 4);
    int colTile = n >> 4;
    int idx = (((chunk * 8 + colTile) * 32 + lane) << 4) + v * 2 + h;
    Bsw[idx] = (_Float16)acc;
}

// H[65536,128] f32 -> f16 in WMMA 16-bit A layout.
// Hsw dword index: ((rowTile*4 + chunk)*32 + lane)*8 + v
__global__ void k_Hsw(const float* __restrict__ H, unsigned int* Hsw) {
    unsigned int idx = blockIdx.x * 256u + threadIdx.x;   // 4,194,304 dwords
    unsigned int v    = idx & 7u;
    unsigned int lane = (idx >> 3) & 31u;
    unsigned int c    = (idx >> 8) & 3u;
    unsigned int rt   = idx >> 10;
    unsigned int m      = lane & 15u;
    unsigned int laneHi = lane >> 4;
    unsigned int k0 = (v < 4u) ? (laneHi * 8u + 2u * v)
                               : (16u + laneHi * 8u + 2u * (v - 4u));
    unsigned int K   = c * 32u + k0;                 // always even
    unsigned int row = rt * 16u + m;
    float2 x = *(const float2*)(H + (size_t)row * KK + K);
    half2v p;
    p.x = (_Float16)x.x;
    p.y = (_Float16)x.y;
    ((half2v*)Hsw)[idx] = p;
}

// K_log = -(H @ d2) / eps  via v_wmma_f32_16x16x32_f16.
// One block = 16 output rows; wave w computes the 16x16 tile at colTile w.
__global__ void k_gemmM(const unsigned int* __restrict__ Hsw,
                        const _Float16* __restrict__ Bsw,
                        float* __restrict__ K_log) {
    unsigned int tid  = threadIdx.x;
    unsigned int wave = tid >> 5;          // 0..7 -> colTile
    unsigned int lane = tid & 31u;
    unsigned int rt   = blockIdx.x;        // 0..4095
    f8v acc = {};
    #pragma unroll
    for (unsigned int c = 0; c < 4; ++c) {
        half16v a = *(const half16v*)(Hsw + (size_t)((rt * 4u + c) * 32u + lane) * 8u);
        half16v b = *(const half16v*)(Bsw + ((size_t)((c * 8u + wave) * 32u + lane) << 4));
        acc = __builtin_amdgcn_wmma_f32_16x16x32_f16(
            false, a, false, b, (short)0, acc, false, false);
    }
    unsigned int laneHi = lane >> 4;
    unsigned int n      = lane & 15u;
    #pragma unroll
    for (int r = 0; r < 8; ++r) {
        unsigned int row = rt * 16u + laneHi * 8u + (unsigned)r;
        K_log[(size_t)row * KK + wave * 16u + n] = acc[r] * (-EPS_INV);
    }
}

// ---------------- Sinkhorn iteration kernels ----------------

// u[i] = a_log[b_i] - logsumexp_k(K_log[i,k] + v[b_i,k]); one wave per node.
__global__ void k_u(const float* __restrict__ K_log, const float* __restrict__ v,
                    const int* __restrict__ bidx, const float* __restrict__ a_log_g,
                    float* __restrict__ u) {
    int wave = threadIdx.x >> 5;
    int lane = threadIdx.x & 31;
    int node = blockIdx.x * 8 + wave;
    int b = bidx[node];
    float4 kl = *(const float4*)(K_log + (size_t)node * KK + lane * 4);
    float4 vv = *(const float4*)(v + (size_t)b * KK + lane * 4);
    float x0 = kl.x + vv.x, x1 = kl.y + vv.y, x2 = kl.z + vv.z, x3 = kl.w + vv.w;
    float m = fmaxf(fmaxf(x0, x1), fmaxf(x2, x3));
    #pragma unroll
    for (int off = 16; off > 0; off >>= 1)
        m = fmaxf(m, __shfl_xor(m, off, 32));
    float s = __expf(x0 - m) + __expf(x1 - m) + __expf(x2 - m) + __expf(x3 - m);
    #pragma unroll
    for (int off = 16; off > 0; off >>= 1)
        s += __shfl_xor(s, off, 32);
    if (lane == 0)
        u[node] = a_log_g[b] - (m + __logf(s));
}

// v[b,k] = b_log - seg_logsumexp_i(K_log[i,k] + u[i]); one block per graph,
// 4-way split over the node segment, online logsumexp, LDS merge.
__global__ void k_v(const float* __restrict__ K_log, const float* __restrict__ u,
                    const int* __restrict__ starts, float* __restrict__ v) {
    __shared__ float sm[512], ss[512];
    int b    = blockIdx.x;
    int tid  = threadIdx.x;
    int col  = tid & 127;
    int part = tid >> 7;
    int s0 = starts[b], s1 = starts[b + 1];
    int len = s1 - s0;
    int chunk = (len + 3) >> 2;
    int i0 = s0 + part * chunk;
    int i1 = min(s1, i0 + chunk);
    float m = -INFINITY, s = 0.0f;
    for (int i = i0; i < i1; ++i) {
        float x = K_log[(size_t)i * KK + col] + u[i];
        if (x <= m) {
            s += __expf(x - m);
        } else {
            s = s * __expf(m - x) + 1.0f;
            m = x;
        }
    }
    sm[tid] = m; ss[tid] = s;
    __syncthreads();
    if (tid < 128) {
        float M = fmaxf(fmaxf(sm[col], sm[col + 128]),
                        fmaxf(sm[col + 256], sm[col + 384]));
        float Ms = isfinite(M) ? M : 0.0f;   // empty-graph guard (matches ref)
        float S = ss[col]       * __expf(sm[col]       - Ms)
                + ss[col + 128] * __expf(sm[col + 128] - Ms)
                + ss[col + 256] * __expf(sm[col + 256] - Ms)
                + ss[col + 384] * __expf(sm[col + 384] - Ms);
        v[b * KK + col] = B_LOG - (Ms + __logf(fmaxf(S, 1e-38f)));
    }
}

// hist[b,k] = normalize_k( sum_i exp(u[i] + v[b,k] + K_log[i,k]) ); empty -> 1/K
__global__ void k_hist(const float* __restrict__ K_log, const float* __restrict__ u,
                       const float* __restrict__ v, const int* __restrict__ starts,
                       const int* __restrict__ cnt, float* __restrict__ hist) {
    __shared__ float ps[512];
    __shared__ float hval[128], red[128];
    int b    = blockIdx.x;
    int tid  = threadIdx.x;
    int col  = tid & 127;
    int part = tid >> 7;
    int s0 = starts[b], s1 = starts[b + 1];
    int len = s1 - s0;
    int chunk = (len + 3) >> 2;
    int i0 = s0 + part * chunk;
    int i1 = min(s1, i0 + chunk);
    float vb = v[b * KK + col];
    float acc = 0.0f;
    for (int i = i0; i < i1; ++i)
        acc += __expf(u[i] + vb + K_log[(size_t)i * KK + col]);
    ps[tid] = acc;
    __syncthreads();
    if (tid < 128) {
        float t = ps[tid] + ps[tid + 128] + ps[tid + 256] + ps[tid + 384];
        hval[tid] = t;
        red[tid]  = t;
    }
    __syncthreads();
    for (int off = 64; off > 0; off >>= 1) {
        if (tid < off) red[tid] += red[tid + off];
        __syncthreads();
    }
    if (tid < 128) {
        float tot = red[0];
        hist[b * KK + tid] = (cnt[b] > 0) ? hval[tid] / (tot + TINYF)
                                          : (1.0f / (float)KK);
    }
}

// embs = hist @ codebook  (tiny fp32 GEMV per graph)
__global__ void k_embs(const float* __restrict__ hist,
                       const float* __restrict__ codebook,
                       float* __restrict__ embs) {
    __shared__ float h[KK];
    int b = blockIdx.x;
    int d = threadIdx.x;
    if (d < KK) h[d] = hist[b * KK + d];
    __syncthreads();
    float acc = 0.0f;
    for (int k = 0; k < KK; ++k)
        acc += h[k] * codebook[k * DD + d];
    embs[b * DD + d] = acc;
}

// ---------------- host launch ----------------

extern "C" void kernel_launch(void* const* d_in, const int* in_sizes, int n_in,
                              void* d_out, int out_size, void* d_ws, size_t ws_size,
                              hipStream_t stream) {
    const float* H        = (const float*)d_in[0];   // [65536,128]
    const int*   bidx     = (const int*)d_in[1];     // [65536]
    const float* codebook = (const float*)d_in[2];   // [128,256]
    (void)in_sizes; (void)n_in; (void)out_size; (void)ws_size;

    float* hist_out = (float*)d_out;                     // 128*128
    float* embs_out = (float*)d_out + BB * KK;           // 128*256

    char* ws = (char*)d_ws;                              // ~50.7 MB used
    float*        K_log  = (float*)(ws + 0);                       // 32 MB
    unsigned int* Hsw    = (unsigned int*)(ws + 33554432);         // 16 MB
    _Float16*     Bsw    = (_Float16*)(ws + 50331648);             // 32 KB
    float*        u      = (float*)(ws + 50364416);                // 256 KB
    float*        v      = (float*)(ws + 50626560);                // 64 KB
    int*          cnt    = (int*)(ws + 50692096);                  // 512 B
    int*          starts = (int*)(ws + 50692608);                  // 129 ints
    float*        a_log  = (float*)(ws + 50693632);                // 512 B

    k_init <<<64, 256, 0, stream>>>(v, cnt);
    k_count<<<NN / 256, 256, 0, stream>>>(bidx, cnt);
    k_scan <<<1, 128, 0, stream>>>(cnt, starts, a_log);
    k_d2sw <<<KK, KK, 0, stream>>>(codebook, Bsw);
    k_Hsw  <<<16384, 256, 0, stream>>>(H, Hsw);
    k_gemmM<<<NN / 16, 256, 0, stream>>>(Hsw, Bsw, K_log);

    for (int it = 0; it < MAXIT; ++it) {
        k_u<<<NN / 8, 256, 0, stream>>>(K_log, v, bidx, a_log, u);
        k_v<<<BB, 512, 0, stream>>>(K_log, u, starts, v);
    }

    k_hist<<<BB, 512, 0, stream>>>(K_log, u, v, starts, cnt, hist_out);
    k_embs<<<BB, DD, 0, stream>>>(hist_out, codebook, embs_out);
}
